// Seq2seqRNN_32573031973387
// MI455X (gfx1250) — matrix operand order repaired and checked by reading, not compiled, not run on hardware
//
#include <hip/hip_runtime.h>
#include <stdint.h>

// ---- problem constants (match reference) ----
#define V_ 32000
#define E_ 256
#define H_ 512
#define B_ 32
#define S_ 512
#define T_ 30
#define G3 1536   // 3*H
#define BOS_ 1

typedef __attribute__((ext_vector_type(16))) __bf16 v16bf;
typedef __attribute__((ext_vector_type(8)))  float  v8f;
typedef __attribute__((ext_vector_type(4)))  unsigned int tdm_g0_t;
typedef __attribute__((ext_vector_type(8)))  int          tdm_g1_t;
typedef __attribute__((ext_vector_type(4)))  int          tdm_g4_t;
typedef __attribute__((ext_vector_type(8)))  int          tdm_g8_t;

union FragAB { uint4 q[2]; v16bf v; };

__device__ __forceinline__ uint16_t f2bf(float f) {
    union { float f; uint32_t u; } x; x.f = f;
    uint32_t r = (x.u + 0x7FFFu + ((x.u >> 16) & 1u)) >> 16;
    return (uint16_t)r;
}
__device__ __forceinline__ float bf2f(uint16_t b) {
    union { uint32_t u; float f; } x; x.u = ((uint32_t)b) << 16;
    return x.f;
}
__device__ __forceinline__ float sigmf(float x) { return 1.0f / (1.0f + __expf(-x)); }

// A fragment (16x32 bf16), row-major source, ld in elements.
__device__ __forceinline__ v16bf ldA(const uint16_t* A, int lda, int m0, int k0, int lane) {
    int m = m0 + (lane & 15);
    int h = lane >> 4;
    const uint16_t* p = A + (size_t)m * lda + k0 + h * 8;
    FragAB f;
    f.q[0] = *(const uint4*)(p);
    f.q[1] = *(const uint4*)(p + 16);
    return f.v;
}
// B fragment (32x16 bf16) taken from row-major W[N][K] (B = W^T).
__device__ __forceinline__ v16bf ldB(const uint16_t* W, int ldw, int ntile, int k0, int lane) {
    int n  = ntile * 16 + (lane & 15);
    int kb = k0 + (lane >> 4) * 8;
    const uint16_t* p = W + (size_t)n * ldw + kb;
    FragAB f;
    f.q[0] = *(const uint4*)(p);
    f.q[1] = *(const uint4*)(p + 16);
    return f.v;
}
__device__ __forceinline__ v8f wmma_bf16(v16bf a, v16bf b, v8f c) {
    return __builtin_amdgcn_wmma_f32_16x16x32_bf16(false, a, false, b, (short)0, c, false, false);
}

// ---- Tensor Data Mover: 2D tile (rows x rowElems of bf16) global -> LDS, with LDS row padding.
// padInterval: value v -> pad every 2^(v+1) DWORDs; padAmount: value v -> v+1 DWORDs of pad.
__device__ __forceinline__ void tdm_load_2d_bf16(const void* gptr, uint32_t lds_off,
                                                 uint32_t rowElems, uint32_t rows,
                                                 uint32_t padInterval, uint32_t padAmount) {
#if defined(__HIP_DEVICE_COMPILE__)
    uint64_t ga = (uint64_t)(uintptr_t)gptr;
    tdm_g0_t g0;
    g0[0] = 1u;                                                    // count=1 valid descriptor
    g0[1] = lds_off;                                               // D#.lds_addr (bytes)
    g0[2] = (uint32_t)ga;                                          // global_addr[31:0]
    g0[3] = (uint32_t)((ga >> 32) & 0x01FFFFFFu) | 0x80000000u;    // addr[56:32] | type=2
    tdm_g1_t g1;
    g1[0] = (int)((1u << 16) | (1u << 20) |                        // data_size=2B, pad_enable
                  (padInterval << 22) | (padAmount << 25));
    g1[1] = (int)((rowElems & 0xFFFFu) << 16);                     // tensor_dim0[15:0]
    g1[2] = (int)(((rowElems >> 16) & 0xFFFFu) |                   // tensor_dim0[31:16]
                  ((rows & 0xFFFFu) << 16));                       // tensor_dim1[15:0]
    g1[3] = (int)(((rows >> 16) & 0xFFFFu) |                       // tensor_dim1[31:16]
                  ((rowElems & 0xFFFFu) << 16));                   // tile_dim0
    g1[4] = (int)(rows & 0xFFFFu);                                 // tile_dim1 (tile_dim2=0)
    g1[5] = (int)rowElems;                                         // tensor_dim0_stride[31:0]
    g1[6] = 0;
    g1[7] = 0;
    tdm_g4_t z4 = {0, 0, 0, 0};
    tdm_g8_t z8 = {0, 0, 0, 0, 0, 0, 0, 0};
    __builtin_amdgcn_tensor_load_to_lds(g0, g1, z4, z4, z8, 0);    // 6-arg (clang-23) form
#endif
}
__device__ __forceinline__ void tdm_wait0() {
#if defined(__HIP_DEVICE_COMPILE__)
    __builtin_amdgcn_s_wait_tensorcnt(0);
#endif
}

// monotonic packed key for argmax: high 32 = order-preserving float bits, low 32 = ~idx (ties -> lowest idx)
__device__ __forceinline__ unsigned long long packkey(float v, int idx) {
    union { float f; unsigned u; } x; x.f = v;
    unsigned k = (x.u & 0x80000000u) ? ~x.u : (x.u | 0x80000000u);
    return ((unsigned long long)k << 32) | (unsigned)(~(unsigned)idx);
}
__device__ __forceinline__ unsigned long long shflx64_16(unsigned long long v, int m) {
    unsigned lo = (unsigned)v, hi = (unsigned)(v >> 32);
    lo = __shfl_xor(lo, m, 16);
    hi = __shfl_xor(hi, m, 16);
    return (((unsigned long long)hi) << 32) | lo;
}

// ---------------- prep kernels ----------------
__global__ void k_convert(const float* __restrict__ src, uint16_t* __restrict__ dst, int n) {
    int i = blockIdx.x * 256 + threadIdx.x;
    if (i < n) dst[i] = f2bf(src[i]);
}

// x_enc[(s*32+b)*256+e] = bf16(enc_emb[body[b,s], e])
__global__ void k_gather_x(const int* __restrict__ body, const float* __restrict__ emb,
                           uint16_t* __restrict__ x) {
    int i = blockIdx.x * 256 + threadIdx.x;
    if (i >= S_ * B_ * E_) return;
    int e  = i & (E_ - 1);
    int sb = i >> 8;
    int b  = sb & 31;
    int s  = sb >> 5;
    int tok = body[b * S_ + s];
    x[i] = f2bf(emb[(size_t)tok * E_ + e]);
}

// gi[s][b][n] = bf16( x[s] @ Wih^T + bih ) : grid = S, block = 256 (8 waves x 12 n-tiles)
// M-tile loop outermost: only 8 A-fragments (64 VGPRs) live per pass -> no scratch spills.
__global__ void k_gi(const uint16_t* __restrict__ xenc, const uint16_t* __restrict__ Wih,
                     const float* __restrict__ bih, uint16_t* __restrict__ gi) {
    __shared__ uint16_t Ax[32 * (E_ + 16)];
    const int LDX = E_ + 16;
    int s = blockIdx.x;
    int tid = threadIdx.x, lane = tid & 31, wave = tid >> 5;
    const uint16_t* xs = xenc + (size_t)s * B_ * E_;
    if (tid < 32) {   // wave 0: TDM DMA 32x256 bf16 tile into padded LDS rows (128 DW + 8 DW pad)
        tdm_load_2d_bf16(xs, (uint32_t)(uintptr_t)(&Ax[0]), E_, 32, /*interval 128DW*/6, /*8DW*/7);
        tdm_wait0();
    }
    __syncthreads();
    int l15 = lane & 15, half = lane >> 4;
    uint16_t* out = gi + (size_t)s * B_ * G3;
    for (int mt = 0; mt < 2; ++mt) {
        for (int i = 0; i < 12; ++i) {
            int nt = wave * 12 + i;
            v8f c = {};
            for (int ks = 0; ks < E_ / 32; ++ks) {
                v16bf b = ldB(Wih, E_, nt, ks * 32, lane);
                v16bf a = ldA(Ax, LDX, mt * 16, ks * 32, lane);
                c = wmma_bf16(a, b, c);
            }
            int col = nt * 16 + l15;
            float bb = bih[col];
#pragma unroll
            for (int r = 0; r < 8; ++r) {
                int m = mt * 16 + r + 8 * half;
                out[(size_t)m * G3 + col] = f2bf(c[r] + bb);
            }
        }
    }
}

// ---------------- encoder recurrence: 1 WG per direction, 512 steps in-kernel ----------------
__global__ void k_enc_rnn(const uint16_t* __restrict__ gi_f, const uint16_t* __restrict__ gi_b,
                          const uint16_t* __restrict__ Whh_f, const uint16_t* __restrict__ Whh_b,
                          const float* __restrict__ bhh_f, const float* __restrict__ bhh_b,
                          float* __restrict__ henc) {
    extern __shared__ char smem[];
    const int LDH = H_ + 16;
    uint16_t* hbf = (uint16_t*)smem;                 // [32][LDH] bf16 h (A-matrix source)
    float*    hf  = (float*)(smem + 32 * LDH * 2);   // [32][H] f32 h state

    int dir = blockIdx.x;
    const uint16_t* gi  = dir ? gi_b  : gi_f;
    const uint16_t* Whh = dir ? Whh_b : Whh_f;
    const float*    bhh = dir ? bhh_b : bhh_f;

    int tid = threadIdx.x, lane = tid & 31, wave = tid >> 5;   // 16 waves
    for (int i = tid; i < 32 * LDH; i += 512) hbf[i] = 0;
    for (int i = tid; i < 32 * H_;  i += 512) hf[i]  = 0.0f;
    __syncthreads();

    int l15 = lane & 15, half = lane >> 4;
    int tj[2] = { wave, wave + 16 };                 // gate-aligned tile pairs
    float bhr[2], bhz[2], bhn[2];
    for (int j = 0; j < 2; ++j) {
        bhr[j] = bhh[tj[j] * 16 + l15];
        bhz[j] = bhh[(32 + tj[j]) * 16 + l15];
        bhn[j] = bhh[(64 + tj[j]) * 16 + l15];
    }

    for (int s = 0; s < S_; ++s) {
        int seff = dir ? (S_ - 1 - s) : s;
        // prefetch next step's gi slice for this thread's columns (hides the only fresh
        // global traffic on the 512-step sequential path behind the WMMA phase)
        if (s + 1 < S_) {
            int snext = dir ? (S_ - 2 - s) : (s + 1);
            const uint16_t* gn = gi + (size_t)snext * B_ * G3 + (size_t)(8 * half) * G3 + tj[0] * 16 + l15;
            __builtin_prefetch(gn, 0, 1);
            __builtin_prefetch(gn + 1024, 0, 1);
        }
        v8f Cr[2][2] = {}, Cz[2][2] = {}, Cn[2][2] = {};
        for (int ks = 0; ks < H_ / 32; ++ks) {
            v16bf a0 = ldA(hbf, LDH, 0, ks * 32, lane);
            v16bf a1 = ldA(hbf, LDH, 16, ks * 32, lane);
            for (int j = 0; j < 2; ++j) {
                v16bf br = ldB(Whh, H_, tj[j], ks * 32, lane);
                Cr[j][0] = wmma_bf16(a0, br, Cr[j][0]);
                Cr[j][1] = wmma_bf16(a1, br, Cr[j][1]);
                v16bf bz = ldB(Whh, H_, 32 + tj[j], ks * 32, lane);
                Cz[j][0] = wmma_bf16(a0, bz, Cz[j][0]);
                Cz[j][1] = wmma_bf16(a1, bz, Cz[j][1]);
                v16bf bn = ldB(Whh, H_, 64 + tj[j], ks * 32, lane);
                Cn[j][0] = wmma_bf16(a0, bn, Cn[j][0]);
                Cn[j][1] = wmma_bf16(a1, bn, Cn[j][1]);
            }
        }
        __syncthreads();   // all waves done reading hbf for this step
        const uint16_t* gis = gi + (size_t)seff * B_ * G3;
        for (int j = 0; j < 2; ++j) {
            int c = tj[j] * 16 + l15;     // hidden column 0..511
#pragma unroll
            for (int mt = 0; mt < 2; ++mt) {
#pragma unroll
                for (int r = 0; r < 8; ++r) {
                    int m = mt * 16 + r + 8 * half;
                    const uint16_t* gp = gis + (size_t)m * G3 + c;
                    float ir  = bf2f(gp[0]);
                    float iz  = bf2f(gp[512]);
                    float in_ = bf2f(gp[1024]);
                    float rr = sigmf(ir + Cr[j][mt][r] + bhr[j]);
                    float zz = sigmf(iz + Cz[j][mt][r] + bhz[j]);
                    float nn = tanhf(in_ + rr * (Cn[j][mt][r] + bhn[j]));
                    float ho = hf[m * H_ + c];
                    float hn2 = (1.0f - zz) * nn + zz * ho;
                    hf[m * H_ + c] = hn2;
                    hbf[m * LDH + c] = f2bf(hn2);
                }
            }
        }
        __syncthreads();
    }
    for (int i = tid; i < 32 * H_; i += 512) henc[(size_t)dir * B_ * H_ + i] = hf[i];
}

// ---------------- decoder init: hidden = h_fwd + h_bwd ----------------
__global__ void k_combine(const float* __restrict__ henc, float* __restrict__ h32,
                          uint16_t* __restrict__ h16, unsigned long long* __restrict__ amax) {
    int i = blockIdx.x * 256 + threadIdx.x;
    if (i < B_ * H_) {
        float h = henc[i] + henc[B_ * H_ + i];
        h32[i] = h;
        h16[i] = f2bf(h);
    }
    if (blockIdx.x == 0 && threadIdx.x < B_) amax[threadIdx.x] = 0ull;
}

// ---------------- decoder GRU cell: 1 WG, 16 waves, gate-pair loop outermost ----------------
__global__ void k_dec_cell(int step, unsigned long long* __restrict__ amax,
                           const float* __restrict__ dec_emb,
                           const uint16_t* __restrict__ Wih, const uint16_t* __restrict__ Whh,
                           const float* __restrict__ bih, const float* __restrict__ bhh,
                           float* __restrict__ h32, uint16_t* __restrict__ h16) {
    __shared__ uint16_t Ax[32 * (E_ + 16)];
    __shared__ uint16_t Ah[32 * (H_ + 16)];
    const int LDX = E_ + 16, LDH = H_ + 16;
    int tid = threadIdx.x, lane = tid & 31, wave = tid >> 5;

    for (int i = tid; i < 32 * E_; i += 512) {
        int b = i >> 8, e = i & (E_ - 1);
        int tok = (step == 0) ? BOS_ : (int)(~(unsigned)(amax[b] & 0xffffffffull));
        if (tok < 0 || tok >= V_) tok = 0;
        Ax[b * LDX + e] = f2bf(dec_emb[(size_t)tok * E_ + e]);
    }
    for (int i = tid; i < 32 * H_; i += 512) {
        int b = i >> 9, k = i & (H_ - 1);
        Ah[b * LDH + k] = f2bf(h32[i]);
    }
    __syncthreads();
    if (tid < B_) amax[tid] = 0ull;   // reset for this step's logits/argmax

    int l15 = lane & 15, half = lane >> 4;
    for (int j = 0; j < 2; ++j) {     // process one gate-aligned tile triple at a time (64 accum VGPRs)
        int tjj = wave + 16 * j;
        v8f Cr[2] = {}, Cz[2] = {}, Cni[2] = {}, Cnh[2] = {};
        // input-side GEMM (K=256): r,z accumulate; n kept separate (Cni)
        for (int ks = 0; ks < E_ / 32; ++ks) {
            v16bf a0 = ldA(Ax, LDX, 0, ks * 32, lane);
            v16bf a1 = ldA(Ax, LDX, 16, ks * 32, lane);
            v16bf br = ldB(Wih, E_, tjj, ks * 32, lane);
            Cr[0] = wmma_bf16(a0, br, Cr[0]);
            Cr[1] = wmma_bf16(a1, br, Cr[1]);
            v16bf bz = ldB(Wih, E_, 32 + tjj, ks * 32, lane);
            Cz[0] = wmma_bf16(a0, bz, Cz[0]);
            Cz[1] = wmma_bf16(a1, bz, Cz[1]);
            v16bf bn = ldB(Wih, E_, 64 + tjj, ks * 32, lane);
            Cni[0] = wmma_bf16(a0, bn, Cni[0]);
            Cni[1] = wmma_bf16(a1, bn, Cni[1]);
        }
        // hidden-side GEMM (K=512): r,z continue accumulating; n separate (Cnh)
        for (int ks = 0; ks < H_ / 32; ++ks) {
            v16bf a0 = ldA(Ah, LDH, 0, ks * 32, lane);
            v16bf a1 = ldA(Ah, LDH, 16, ks * 32, lane);
            v16bf br = ldB(Whh, H_, tjj, ks * 32, lane);
            Cr[0] = wmma_bf16(a0, br, Cr[0]);
            Cr[1] = wmma_bf16(a1, br, Cr[1]);
            v16bf bz = ldB(Whh, H_, 32 + tjj, ks * 32, lane);
            Cz[0] = wmma_bf16(a0, bz, Cz[0]);
            Cz[1] = wmma_bf16(a1, bz, Cz[1]);
            v16bf bn = ldB(Whh, H_, 64 + tjj, ks * 32, lane);
            Cnh[0] = wmma_bf16(a0, bn, Cnh[0]);
            Cnh[1] = wmma_bf16(a1, bn, Cnh[1]);
        }
        int c = tjj * 16 + l15;
        float br_ = bih[c] + bhh[c];
        float bz_ = bih[512 + c] + bhh[512 + c];
        float bin = bih[1024 + c];
        float bhn = bhh[1024 + c];
#pragma unroll
        for (int mt = 0; mt < 2; ++mt) {
#pragma unroll
            for (int r = 0; r < 8; ++r) {
                int m = mt * 16 + r + 8 * half;
                float rr = sigmf(Cr[mt][r] + br_);
                float zz = sigmf(Cz[mt][r] + bz_);
                float nn = tanhf(Cni[mt][r] + bin + rr * (Cnh[mt][r] + bhn));
                float ho = h32[m * H_ + c];
                float hn2 = (1.0f - zz) * nn + zz * ho;
                h32[m * H_ + c] = hn2;
                h16[m * H_ + c] = f2bf(hn2);
            }
        }
    }
}

// ---------------- logits + fused argmax: grid 50 x 8 waves x 5 tiles = 2000 N-tiles ----------------
// M-tile loop outermost: 16 live A-fragments per pass instead of 32.
__global__ void k_logits(int t, const uint16_t* __restrict__ h16,
                         const uint16_t* __restrict__ fcW, const float* __restrict__ fcb,
                         float* __restrict__ out, unsigned long long* __restrict__ amax) {
    __shared__ uint16_t Ah[32 * (H_ + 16)];
    const int LDH = H_ + 16;
    int tid = threadIdx.x, lane = tid & 31, wave = tid >> 5;
    if (tid < 32) {   // wave 0: TDM DMA 32x512 bf16 h tile into padded LDS rows (256 DW + 8 DW pad)
        tdm_load_2d_bf16(h16, (uint32_t)(uintptr_t)(&Ah[0]), H_, 32, /*interval 256DW*/7, /*8DW*/7);
        tdm_wait0();
    }
    __syncthreads();
    int l15 = lane & 15, half = lane >> 4;
    for (int mt = 0; mt < 2; ++mt) {
        for (int i = 0; i < 5; ++i) {
            int nt = (blockIdx.x * 8 + wave) * 5 + i;
            v8f c = {};
            for (int ks = 0; ks < H_ / 32; ++ks) {
                v16bf b = ldB(fcW, H_, nt, ks * 32, lane);
                v16bf a = ldA(Ah, LDH, mt * 16, ks * 32, lane);
                c = wmma_bf16(a, b, c);
            }
            int col = nt * 16 + l15;
            float bb = fcb[col];
            unsigned long long pk[8];
#pragma unroll
            for (int r = 0; r < 8; ++r) {
                int m = mt * 16 + r + 8 * half;
                float v = c[r] + bb;
                out[(size_t)m * (T_ * V_) + (size_t)t * V_ + col] = v;
                pk[r] = packkey(v, col);
            }
#pragma unroll
            for (int r = 0; r < 8; ++r) {
                unsigned long long p = pk[r];
                for (int ms = 1; ms < 16; ms <<= 1) {
                    unsigned long long q = shflx64_16(p, ms);
                    if (q > p) p = q;
                }
                if (l15 == 0) {
                    int m = mt * 16 + r + 8 * half;
                    atomicMax(&amax[m], p);
                }
            }
        }
    }
}

// ---------------- host orchestration ----------------
extern "C" void kernel_launch(void* const* d_in, const int* in_sizes, int n_in,
                              void* d_out, int out_size, void* d_ws, size_t ws_size,
                              hipStream_t stream) {
    (void)in_sizes; (void)n_in; (void)out_size; (void)ws_size;
    const int*   body      = (const int*)d_in[0];
    const float* enc_emb   = (const float*)d_in[2];
    const float* enc_wih_f = (const float*)d_in[3];
    const float* enc_whh_f = (const float*)d_in[4];
    const float* enc_bih_f = (const float*)d_in[5];
    const float* enc_bhh_f = (const float*)d_in[6];
    const float* enc_wih_b = (const float*)d_in[7];
    const float* enc_whh_b = (const float*)d_in[8];
    const float* enc_bih_b = (const float*)d_in[9];
    const float* enc_bhh_b = (const float*)d_in[10];
    const float* dec_emb   = (const float*)d_in[11];
    const float* dec_wih   = (const float*)d_in[12];
    const float* dec_whh   = (const float*)d_in[13];
    const float* dec_bih   = (const float*)d_in[14];
    const float* dec_bhh   = (const float*)d_in[15];
    const float* fc_w      = (const float*)d_in[16];
    const float* fc_b      = (const float*)d_in[17];
    float* out = (float*)d_out;

    char* ws = (char*)d_ws;
    size_t off = 0;
    auto alloc = [&](size_t bytes) -> char* {
        char* p = ws + off;
        off = (off + bytes + 255) & ~(size_t)255;
        return p;
    };
    uint16_t* wWihF = (uint16_t*)alloc((size_t)G3 * E_ * 2);
    uint16_t* wWhhF = (uint16_t*)alloc((size_t)G3 * H_ * 2);
    uint16_t* wWihB = (uint16_t*)alloc((size_t)G3 * E_ * 2);
    uint16_t* wWhhB = (uint16_t*)alloc((size_t)G3 * H_ * 2);
    uint16_t* wDWih = (uint16_t*)alloc((size_t)G3 * E_ * 2);
    uint16_t* wDWhh = (uint16_t*)alloc((size_t)G3 * H_ * 2);
    uint16_t* wFc   = (uint16_t*)alloc((size_t)V_ * H_ * 2);
    uint16_t* xenc  = (uint16_t*)alloc((size_t)S_ * B_ * E_ * 2);
    uint16_t* giF   = (uint16_t*)alloc((size_t)S_ * B_ * G3 * 2);
    uint16_t* giB   = (uint16_t*)alloc((size_t)S_ * B_ * G3 * 2);
    float*    henc  = (float*)alloc((size_t)2 * B_ * H_ * 4);
    float*    h32   = (float*)alloc((size_t)B_ * H_ * 4);
    uint16_t* h16   = (uint16_t*)alloc((size_t)B_ * H_ * 2);
    unsigned long long* amax = (unsigned long long*)alloc(B_ * 8);

    auto cvt = [&](const float* s, uint16_t* d, int n) {
        k_convert<<<(n + 255) / 256, 256, 0, stream>>>(s, d, n);
    };
    cvt(enc_wih_f, wWihF, G3 * E_);
    cvt(enc_whh_f, wWhhF, G3 * H_);
    cvt(enc_wih_b, wWihB, G3 * E_);
    cvt(enc_whh_b, wWhhB, G3 * H_);
    cvt(dec_wih,   wDWih, G3 * E_);
    cvt(dec_whh,   wDWhh, G3 * H_);
    cvt(fc_w,      wFc,   V_ * H_);

    k_gather_x<<<(S_ * B_ * E_ + 255) / 256, 256, 0, stream>>>(body, enc_emb, xenc);
    k_gi<<<S_, 256, 0, stream>>>(xenc, wWihF, enc_bih_f, giF);
    k_gi<<<S_, 256, 0, stream>>>(xenc, wWihB, enc_bih_b, giB);

    const int LDH = H_ + 16;
    size_t rnn_lds = (size_t)32 * LDH * 2 + (size_t)32 * H_ * 4;   // 99,328 B dynamic LDS (of 320 KB/WGP)
    k_enc_rnn<<<2, 512, rnn_lds, stream>>>(giF, giB, wWhhF, wWhhB, enc_bhh_f, enc_bhh_b, henc);

    k_combine<<<(B_ * H_ + 255) / 256, 256, 0, stream>>>(henc, h32, h16, amax);

    for (int t = 0; t < T_; ++t) {
        k_dec_cell<<<1, 512, 0, stream>>>(t, amax, dec_emb, wDWih, wDWhh,
                                          dec_bih, dec_bhh, h32, h16);
        k_logits<<<50, 256, 0, stream>>>(t, h16, wFc, fc_b, out, amax);
    }
}
